// PredictiveCodingLayer_26336739459342
// MI455X (gfx1250) — matrix operand order repaired and checked by reading, not compiled, not run on hardware
//
#include <hip/hip_runtime.h>

typedef __bf16 bf16_t;
typedef __attribute__((ext_vector_type(16))) __bf16 v16bf;
typedef __attribute__((ext_vector_type(8)))  __bf16 v8bf;
typedef __attribute__((ext_vector_type(4)))  __bf16 v4bf;
typedef __attribute__((ext_vector_type(8)))  float  v8f;
typedef __attribute__((ext_vector_type(4)))  float  v4f;

#define B_ROWS 16384
#define DDIM   2048
#define KSTEP  32
#define NKT    (DDIM / KSTEP)

enum { EPI_REC = 0, EPI_ERR = 1, EPI_GRAD = 2, EPI_FINAL = 3 };

// ---------------------------------------------------------------------------
// Async global->LDS copy of one 16-byte chunk (per lane) via the CDNA5
// GLOBAL_LOAD_ASYNC_TO_LDS_B128 path (ASYNCcnt-tracked, no staging VGPRs).
// Builtin signature (from hipcc diagnostic): (int4 AS1*, int4 AS3*, Ii, Ii).
// ---------------------------------------------------------------------------
typedef __attribute__((__vector_size__(16))) int i4v;
typedef __attribute__((address_space(1))) i4v* gp_i4;
typedef __attribute__((address_space(3))) i4v* lp_i4;
typedef __attribute__((address_space(3))) bf16_t lds_bf16;

__device__ __forceinline__ void async_cp16(const bf16_t* g, bf16_t* l) {
#if __has_builtin(__builtin_amdgcn_global_load_async_to_lds_b128)
  __builtin_amdgcn_global_load_async_to_lds_b128((gp_i4)g, (lp_i4)l, 0, 0);
#else
  unsigned loff = (unsigned)(size_t)(lds_bf16*)l;        // LDS byte offset
  unsigned long long ga = (unsigned long long)(size_t)g; // 64-bit global addr
  asm volatile("global_load_async_to_lds_b128 %0, %1, off"
               :: "v"(loff), "v"(ga) : "memory");
#endif
}

__device__ __forceinline__ void async_wait0() {
#if __has_builtin(__builtin_amdgcn_s_wait_asynccnt)
  __builtin_amdgcn_s_wait_asynccnt(0);
#else
  asm volatile("s_wait_asynccnt 0x0" ::: "memory");
#endif
}

// ---------------------------------------------------------------------------
// WMMA bf16 GEMM: C[16384 x 2048] = A[16384 x K] * B^T, Bw stored [N][K].
// Block tile 128(M) x 256(N), 8 waves (2x4), each wave 64x64 = 4x4 wmma tiles.
// LDS double-buffered; tiles arrive via GLOBAL_LOAD_ASYNC_TO_LDS_B128.
// ---------------------------------------------------------------------------
template <int MODE>
__global__ __launch_bounds__(256) void gemm_bf16(
    const bf16_t* __restrict__ A,       // [16384 x 2048] M x K
    const bf16_t* __restrict__ Bw,      // [2048 x 2048]  N x K
    const float*  __restrict__ bias,    // [N]   (REC/ERR/FINAL)
    const float*  __restrict__ xin,     // x     (ERR/FINAL)
    const float*  __restrict__ p2,      // [N]   (ERR)
    const float*  __restrict__ hid_in,  // hidden fp32 (GRAD)
    float*        __restrict__ outF,    // fp32 out (REC/GRAD/FINAL)
    bf16_t*       __restrict__ outB,    // bf16 out (ERR)
    float*        __restrict__ accSlot, // scalar |err| accumulator
    int track, float scale)
{
  // per buffer: A tile [128][32] (4096) then B tile [256][32] (8192)
  __shared__ __align__(64) bf16_t smem[2][12288];

  const int t    = threadIdx.x;
  const int lane = t & 31;
  const int w    = t >> 5;
  const int wm   = w >> 2;   // 0..1  (M direction)
  const int wn   = w & 3;    // 0..3  (N direction)
  const int lrow = lane & 15;
  const int lhi  = lane >> 4;
  const int n0   = blockIdx.x * 256;
  const int m0   = blockIdx.y * 128;
  const int K    = DDIM;

  v8f acc[4][4];
#pragma unroll
  for (int s = 0; s < 4; ++s)
#pragma unroll
    for (int u = 0; u < 4; ++u)
#pragma unroll
      for (int r = 0; r < 8; ++r) acc[s][u][r] = 0.0f;

  // ---- issue async stage of one K-tile into LDS buffer `buf`
  auto stage = [&](int buf, int kt) {
    bf16_t* sA = &smem[buf][0];
    bf16_t* sB = &smem[buf][4096];
#pragma unroll
    for (int i = 0; i < 2; ++i) {          // A: 512 x 16B chunks
      const int c = t + i * 256;
      const int row = c >> 2;
      const int off = (c & 3) * 8;
      async_cp16(A + (size_t)(m0 + row) * K + kt * KSTEP + off,
                 &sA[row * KSTEP + off]);
    }
#pragma unroll
    for (int i = 0; i < 4; ++i) {          // B: 1024 x 16B chunks
      const int c = t + i * 256;
      const int row = c >> 2;
      const int off = (c & 3) * 8;
      async_cp16(Bw + (size_t)(n0 + row) * K + kt * KSTEP + off,
                 &sB[row * KSTEP + off]);
    }
  };

  stage(0, 0);
  async_wait0();
  __syncthreads();

  for (int kt = 0; kt < NKT; ++kt) {
    const int cur = kt & 1;
    if (kt + 1 < NKT) stage(1 - cur, kt + 1);  // DMA next tile during compute

    const bf16_t* sA = &smem[cur][0];
    const bf16_t* sB = &smem[cur][4096];

    // A fragments (16-bit A 16x32 layout: per lane two 16B chunks)
    v16bf af[4];
#pragma unroll
    for (int s = 0; s < 4; ++s) {
      const int m = wm * 64 + s * 16 + lrow;
      v8bf lo = *reinterpret_cast<const v8bf*>(&sA[m * KSTEP + lhi * 16]);
      v8bf hi = *reinterpret_cast<const v8bf*>(&sA[m * KSTEP + lhi * 16 + 8]);
      af[s] = __builtin_shufflevector(lo, hi, 0, 1, 2, 3, 4, 5, 6, 7,
                                      8, 9, 10, 11, 12, 13, 14, 15);
    }
    // B fragments (lane = column n, 16 contiguous K starting at 16*lhi)
    v16bf bfr[4];
#pragma unroll
    for (int u = 0; u < 4; ++u) {
      const int n = wn * 64 + u * 16 + lrow;
      bfr[u] = *reinterpret_cast<const v16bf*>(&sB[n * KSTEP + lhi * 16]);
    }

#pragma unroll
    for (int s = 0; s < 4; ++s)
#pragma unroll
      for (int u = 0; u < 4; ++u)
        acc[s][u] = __builtin_amdgcn_wmma_f32_16x16x32_bf16(
            false, af[s], false, bfr[u], (short)0, acc[s][u], false, false);

    async_wait0();     // next-tile DMA complete (in-flight during the WMMAs)
    __syncthreads();
  }

  // ---- epilogue: C layout M = r + 8*lhi, N = lrow within each 16x16 tile
  float labs = 0.0f;
#pragma unroll
  for (int s = 0; s < 4; ++s) {
#pragma unroll
    for (int u = 0; u < 4; ++u) {
      const int mBase = m0 + wm * 64 + s * 16 + 8 * lhi;
      const int nIdx  = n0 + wn * 64 + u * 16 + lrow;
#pragma unroll
      for (int r = 0; r < 8; ++r) {
        const size_t idx = (size_t)(mBase + r) * DDIM + nIdx;
        const float v = acc[s][u][r];
        if (MODE == EPI_REC) {
          outF[idx] = v + bias[nIdx];
        } else if (MODE == EPI_ERR) {
          const float e = xin[idx] - (v + bias[nIdx]);
          outB[idx] = (bf16_t)(e * p2[nIdx]);
          if (track) labs += fabsf(e);
        } else if (MODE == EPI_GRAD) {
          outF[idx] = hid_in[idx] + scale * v;   // hidden + 2*LR*((e*p2)@Wg)
        } else {  // EPI_FINAL
          const float pr = v + bias[nIdx];
          outF[idx] = pr;
          labs += fabsf(xin[idx] - pr);
        }
      }
    }
  }
  if ((MODE == EPI_ERR && track) || MODE == EPI_FINAL) {
#pragma unroll
    for (int o = 16; o > 0; o >>= 1) labs += __shfl_xor(labs, o, 32);
    if (lane == 0) atomicAdd(accSlot, labs);
  }
}

// ---------------------------------------------------------------------------
// Row layernorm over 2048 elems; writes fp32 hidden and bf16 A-operand copy.
// ---------------------------------------------------------------------------
__global__ __launch_bounds__(256) void ln_kernel(
    const float* __restrict__ hpre, const float* __restrict__ gamma,
    const float* __restrict__ beta, float* __restrict__ hout,
    bf16_t* __restrict__ hb)
{
  __shared__ float red[16];
  const int row = blockIdx.x;
  const int t = threadIdx.x;
  const size_t base = (size_t)row * DDIM + t * 8;

  v4f a = *reinterpret_cast<const v4f*>(hpre + base);
  v4f b = *reinterpret_cast<const v4f*>(hpre + base + 4);
  float s = 0.0f, sq = 0.0f;
#pragma unroll
  for (int j = 0; j < 4; ++j) { s += a[j]; sq += a[j] * a[j]; }
#pragma unroll
  for (int j = 0; j < 4; ++j) { s += b[j]; sq += b[j] * b[j]; }
#pragma unroll
  for (int o = 16; o > 0; o >>= 1) {
    s  += __shfl_xor(s,  o, 32);
    sq += __shfl_xor(sq, o, 32);
  }
  if ((t & 31) == 0) { red[t >> 5] = s; red[8 + (t >> 5)] = sq; }
  __syncthreads();
  if (t == 0) {
    float S = 0.0f, Q = 0.0f;
    for (int i = 0; i < 8; ++i) { S += red[i]; Q += red[8 + i]; }
    const float mu  = S * (1.0f / DDIM);
    const float var = Q * (1.0f / DDIM) - mu * mu;
    red[0] = mu;
    red[1] = rsqrtf(var + 1e-5f);
  }
  __syncthreads();
  const float mu = red[0], rs = red[1];

  v4f g0 = *reinterpret_cast<const v4f*>(gamma + t * 8);
  v4f g1 = *reinterpret_cast<const v4f*>(gamma + t * 8 + 4);
  v4f b0 = *reinterpret_cast<const v4f*>(beta + t * 8);
  v4f b1 = *reinterpret_cast<const v4f*>(beta + t * 8 + 4);
  v4f y0, y1;
  v8bf ob;
#pragma unroll
  for (int j = 0; j < 4; ++j) {
    y0[j] = (a[j] - mu) * rs * g0[j] + b0[j];
    y1[j] = (b[j] - mu) * rs * g1[j] + b1[j];
    ob[j]     = (bf16_t)y0[j];
    ob[j + 4] = (bf16_t)y1[j];
  }
  *reinterpret_cast<v4f*>(hout + base)     = y0;
  *reinterpret_cast<v4f*>(hout + base + 4) = y1;
  *reinterpret_cast<v8bf*>(hb + base)      = ob;
}

// ---------------------------------------------------------------------------
// Small helpers
// ---------------------------------------------------------------------------
__global__ void k_cvt(const float* __restrict__ in, bf16_t* __restrict__ out) {
  const size_t i4 = ((size_t)blockIdx.x * 256 + threadIdx.x) * 4;
  v4f v = *reinterpret_cast<const v4f*>(in + i4);
  v4bf o;
#pragma unroll
  for (int j = 0; j < 4; ++j) o[j] = (bf16_t)v[j];
  *reinterpret_cast<v4bf*>(out + i4) = o;
}

__global__ void k_trcvt(const float* __restrict__ in, bf16_t* __restrict__ out) {
  const size_t idx = (size_t)blockIdx.x * 256 + threadIdx.x;  // over D*D
  const int h = (int)(idx >> 11);
  const int i = (int)(idx & (DDIM - 1));
  out[idx] = (bf16_t)in[(size_t)i * DDIM + h];   // out[h][i] = Wg[i][h]
}

__global__ void k_zero(float* accs) {
  if (threadIdx.x < 8) accs[threadIdx.x] = 0.0f;
}

__global__ __launch_bounds__(256) void k_prec(const float* __restrict__ lp,
                                              float* __restrict__ p2,
                                              float* __restrict__ accs) {
  __shared__ float red[8];
  const int t = threadIdx.x;
  float s = 0.0f;
#pragma unroll
  for (int j = 0; j < 8; ++j) {
    const int i = t * 8 + j;
    const float z = lp[i];
    const float sp = (z > 20.0f) ? z : log1pf(expf(z));
    p2[i] = sp * sp;
    s += sp;
  }
#pragma unroll
  for (int o = 16; o > 0; o >>= 1) s += __shfl_xor(s, o, 32);
  if ((t & 31) == 0) red[t >> 5] = s;
  __syncthreads();
  if (t == 0) {
    float S = 0.0f;
    for (int i = 0; i < 8; ++i) S += red[i];
    accs[3] = S;
  }
}

__global__ void k_final(const float* __restrict__ accs, float* __restrict__ scal) {
  const float invBD = 1.0f / ((float)B_ROWS * (float)DDIM);
  scal[0] = accs[2] * invBD;                 // pred_err
  scal[1] = (accs[0] - accs[1]) * invBD;     // err_reduction
  scal[2] = accs[3] * (1.0f / (float)DDIM);  // prec_mean
}

// ---------------------------------------------------------------------------
extern "C" void kernel_launch(void* const* d_in, const int* in_sizes, int n_in,
                              void* d_out, int out_size, void* d_ws, size_t ws_size,
                              hipStream_t stream) {
  (void)in_sizes; (void)n_in; (void)out_size; (void)ws_size;

  const float* x  = (const float*)d_in[0];  // [16384,2048]
  const float* Wg = (const float*)d_in[1];  // [2048,2048]  (D_IN x D_H)
  const float* bg = (const float*)d_in[2];  // [2048]
  const float* Wr = (const float*)d_in[3];  // [2048,2048]  (D_H x D_IN)
  const float* br = (const float*)d_in[4];  // [2048]
  const float* lp = (const float*)d_in[5];  // [2048]
  const float* ga = (const float*)d_in[6];  // [2048]
  const float* be = (const float*)d_in[7];  // [2048]

  const size_t BD = (size_t)B_ROWS * DDIM;
  const size_t DD = (size_t)DDIM * DDIM;

  float* hidden  = (float*)d_out;           // fp32 hidden output region
  float* predbuf = hidden + BD;             // prediction region; doubles as pre-LN scratch
  float* scal    = hidden + 2 * BD;         // 3 scalar outputs

  char* wp = (char*)d_ws;
  bf16_t* xb   = (bf16_t*)wp; wp += BD * 2;
  bf16_t* hb   = (bf16_t*)wp; wp += BD * 2;
  bf16_t* eb   = (bf16_t*)wp; wp += BD * 2;
  bf16_t* Wgb  = (bf16_t*)wp; wp += DD * 2;   // B for prediction GEMM  ([n=i][k=h])
  bf16_t* WgTb = (bf16_t*)wp; wp += DD * 2;   // B for gradient GEMM    ([n=h][k=i])
  bf16_t* Wrb  = (bf16_t*)wp; wp += DD * 2;   // B for recognizer GEMM  ([n=h][k=i])
  float*  p2   = (float*)wp;  wp += DDIM * 4;
  float*  accs = (float*)wp;  wp += 64;

  const dim3 gemmGrid(DDIM / 256, B_ROWS / 128);

  // ---- prep
  k_zero<<<1, 32, 0, stream>>>(accs);
  k_prec<<<1, 256, 0, stream>>>(lp, p2, accs);
  k_cvt<<<(unsigned)(BD / 1024), 256, 0, stream>>>(x, xb);
  k_cvt<<<(unsigned)(DD / 1024), 256, 0, stream>>>(Wg, Wgb);
  k_cvt<<<(unsigned)(DD / 1024), 256, 0, stream>>>(Wr, Wrb);
  k_trcvt<<<(unsigned)(DD / 256), 256, 0, stream>>>(Wg, WgTb);

  // ---- recognizer: hpre = x @ Wr^T + br ; hidden = LN(hpre)
  gemm_bf16<EPI_REC><<<gemmGrid, 256, 0, stream>>>(
      xb, Wrb, br, nullptr, nullptr, nullptr, predbuf, nullptr, nullptr, 0, 0.0f);
  ln_kernel<<<B_ROWS, 256, 0, stream>>>(predbuf, ga, be, hidden, hb);

  // ---- predictive-coding iterations
  for (int it = 0; it < 3; ++it) {
    float* slot = (it == 0) ? (accs + 0) : (accs + 1);
    const int track = (it != 1);
    // prediction + error: eb = bf16((x - (h@Wg^T + bg)) * p2)
    gemm_bf16<EPI_ERR><<<gemmGrid, 256, 0, stream>>>(
        hb, Wgb, bg, x, p2, nullptr, nullptr, eb, slot, track, 0.0f);
    // hpre = hidden + 2*LR * (eb @ Wg)
    gemm_bf16<EPI_GRAD><<<gemmGrid, 256, 0, stream>>>(
        eb, WgTb, nullptr, nullptr, nullptr, hidden, predbuf, nullptr, nullptr, 0, 0.2f);
    ln_kernel<<<B_ROWS, 256, 0, stream>>>(predbuf, ga, be, hidden, hb);
  }

  // ---- final prediction + scalars
  gemm_bf16<EPI_FINAL><<<gemmGrid, 256, 0, stream>>>(
      hb, Wgb, bg, x, nullptr, nullptr, predbuf, nullptr, accs + 2, 1, 0.0f);
  k_final<<<1, 1, 0, stream>>>(accs, scal);
}
